// Generator_66228395704726
// MI455X (gfx1250) — matrix operand order repaired and checked by reading, not compiled, not run on hardware
//
#include <hip/hip_runtime.h>
#include <hip/hip_bf16.h>

// ---------------- problem constants ----------------
#define BB   64
#define ZZ   128
#define CIN  8
#define LL   1024      // 32*32
#define C0C  256
#define CQK_ 32
#define FF   64
#define EPS_ 1e-5f

typedef __attribute__((ext_vector_type(16))) _Float16 v16h;
typedef __attribute__((ext_vector_type(8)))  _Float16 v8h;
typedef __attribute__((ext_vector_type(8)))  float    v8f;

// ---------------- workspace layout (bytes) ----------------
static const size_t OFF_XPRE  = 0;
static const size_t OFF_X0    = OFF_XPRE + (size_t)BB*CIN*LL*4;
static const size_t OFF_X1    = OFF_X0   + (size_t)BB*C0C*LL*4;
static const size_t OFF_Y1    = OFF_X1   + (size_t)BB*C0C*LL*4;
static const size_t OFF_QH    = OFF_Y1   + (size_t)BB*FF*LL*4;
static const size_t OFF_KH    = OFF_QH   + (size_t)BB*LL*CQK_*2;
static const size_t OFF_VH    = OFF_KH   + (size_t)BB*LL*CQK_*2;
static const size_t OFF_STATS = OFF_VH   + (size_t)BB*C0C*LL*2;

// ======================================================================
// 1) x_pre = z @ w_pre^T + b_pre          [64, 8192]
// ======================================================================
__global__ void k_pre(const float* __restrict__ z, const float* __restrict__ wpre,
                      const float* __restrict__ bpre, float* __restrict__ xpre) {
    int idx = blockIdx.x * 256 + threadIdx.x;        // 524288 total
    int b = idx >> 13, r = idx & 8191;
    const float4* zr = (const float4*)(z + (size_t)b * ZZ);
    const float4* wr = (const float4*)(wpre + (size_t)r * ZZ);
    float acc = 0.f;
#pragma unroll 8
    for (int k = 0; k < ZZ / 4; ++k) {
        float4 a = zr[k], w = wr[k];
        acc += a.x * w.x + a.y * w.y + a.z * w.z + a.w * w.w;
    }
    xpre[idx] = acc + bpre[r];
}

// ======================================================================
// 2) conv0 (K=8) + per-channel sum/sumsq partials.  block = (b, o)
// ======================================================================
__global__ void k_conv0(const float* __restrict__ xpre, const float* __restrict__ w0,
                        const float* __restrict__ b0, float* __restrict__ x0,
                        float* __restrict__ sum0, float* __restrict__ sq0) {
    int t = threadIdx.x;
    int o = blockIdx.x & 255, b = blockIdx.x >> 8;
    float wl[CIN];
#pragma unroll
    for (int c = 0; c < CIN; ++c) wl[c] = w0[o * CIN + c];
    float bias = b0[o];
    const float* xb = xpre + (size_t)b * (CIN * LL);
    size_t ob = ((size_t)b * C0C + o) << 10;
    float ls = 0.f, lq = 0.f;
#pragma unroll
    for (int k = 0; k < 4; ++k) {
        int hw = t + 256 * k;
        float y = bias;
#pragma unroll
        for (int c = 0; c < CIN; ++c) y += wl[c] * xb[c * LL + hw];
        x0[ob + hw] = y;
        ls += y; lq += y * y;
    }
    __shared__ float rs[256], rq[256];
    rs[t] = ls; rq[t] = lq; __syncthreads();
    for (int s = 128; s > 0; s >>= 1) {
        if (t < s) { rs[t] += rs[t + s]; rq[t] += rq[t + s]; }
        __syncthreads();
    }
    if (t == 0) { atomicAdd(&sum0[o], rs[0]); atomicAdd(&sq0[o], rq[0]); }
}

// ======================================================================
// 3) finalize BN scale/shift (training-mode, biased var)
// ======================================================================
__global__ void k_bnfin(const float* __restrict__ sum, const float* __restrict__ sq,
                        const float* __restrict__ g, const float* __restrict__ be,
                        float* __restrict__ sc, float* __restrict__ sh,
                        int C, float invN) {
    int c = threadIdx.x;
    if (c < C) {
        float m = sum[c] * invN;
        float v = sq[c] * invN - m * m;
        float s = g[c] * rsqrtf(v + EPS_);
        sc[c] = s;
        sh[c] = be[c] - m * s;
    }
}

// ======================================================================
// 4) BN + ReLU in place on X0
// ======================================================================
__global__ void k_bnrelu(float* __restrict__ x0, const float* __restrict__ sc,
                         const float* __restrict__ sh) {
    size_t base = (size_t)blockIdx.x * 1024 + threadIdx.x;
#pragma unroll
    for (int k = 0; k < 4; ++k) {
        size_t idx = base + 256 * k;
        int c = (int)((idx >> 10) & 255);
        float v = x0[idx] * sc[c] + sh[c];
        x0[idx] = fmaxf(v, 0.f);
    }
}

// ======================================================================
// 5) q/k/v pointwise convs -> f16 buffers in WMMA-friendly layouts.
// ======================================================================
__global__ void __launch_bounds__(256) k_qkv(
        const float* __restrict__ x0,
        const float* __restrict__ wq, const float* __restrict__ bq,
        const float* __restrict__ wk, const float* __restrict__ bk,
        const float* __restrict__ wv, const float* __restrict__ bv,
        _Float16* __restrict__ QH, _Float16* __restrict__ KH,
        _Float16* __restrict__ VH) {
    int jt = blockIdx.x & 63, b = blockIdx.x >> 6;
    int j0 = jt << 4;
    int t = threadIdx.x;
    __shared__ float xt[C0C][16];
    {
        const float* src = x0 + (((size_t)b * C0C + t) << 10) + j0;
#pragma unroll
        for (int jj = 0; jj < 16; ++jj) xt[t][jj] = src[jj];
    }
    __syncthreads();
    // ---- V: channel c = t, 16 columns ----
    {
        float acc[16];
        float bias = bv[t];
#pragma unroll
        for (int jj = 0; jj < 16; ++jj) acc[jj] = bias;
        const float* wr = wv + (size_t)t * C0C;
        for (int cc = 0; cc < C0C; ++cc) {
            float w = wr[cc];
#pragma unroll
            for (int jj = 0; jj < 16; ++jj) acc[jj] += w * xt[cc][jj];
        }
        _Float16* dst = VH + (((size_t)b * C0C + t) << 10) + j0;
#pragma unroll
        for (int jj = 0; jj < 16; ++jj) dst[jj] = (_Float16)acc[jj];
    }
    // ---- Q / K: first 64 threads (32 q channels, 32 k channels) ----
    if (t < 64) {
        int isK = t >> 5, c = t & 31;
        const float* wr = (isK ? wk : wq) + (size_t)c * C0C;
        float bias = isK ? bk[c] : bq[c];
        float acc[16];
#pragma unroll
        for (int jj = 0; jj < 16; ++jj) acc[jj] = bias;
        for (int cc = 0; cc < C0C; ++cc) {
            float w = wr[cc];
#pragma unroll
            for (int jj = 0; jj < 16; ++jj) acc[jj] += w * xt[cc][jj];
        }
        _Float16* base = isK ? KH : QH;
#pragma unroll
        for (int jj = 0; jj < 16; ++jj)
            base[(((size_t)b << 10) + j0 + jj) * CQK_ + c] = (_Float16)acc[jj];
    }
}

// ======================================================================
// 6) Fused attention with WMMA f16 (16x16x32), per 16-row i-tile.
//    grid = B * (L/16) = 4096 blocks of 256 threads (8 waves).
//    P is [i][j] row-major, stride 1032 halves (bank-conflict-free,
//    contiguous B-fragments and softmax rows).
// ======================================================================
#define PSTR 1032   // padded row stride in halves (2064 B -> 516 words, 4*i banks)

__global__ void __launch_bounds__(256) k_attn(
        const _Float16* __restrict__ QH, const _Float16* __restrict__ KH,
        const _Float16* __restrict__ VH, const float* __restrict__ x0,
        const float* __restrict__ gamma, float* __restrict__ x1) {
    const int t = threadIdx.x;
    const int wave = t >> 5, lane = t & 31;
    const int it = blockIdx.x & 63, b = blockIdx.x >> 6;
    const int i0 = it << 4;
    const int ln = lane & 15;
    const bool hiL = lane >= 16;
    const int kb = hiL ? 8 : 0;      // A-fragment K base
    const int mo = hiL ? 8 : 0;      // C/D M offset

    __shared__ _Float16 P[16 * PSTR];          // ~33 KB, [i][j]
    __shared__ float red[256];
    __shared__ float rowmax[16], rowsum[16];

    // ---- A fragment: q tile (M=i rows, K=c). ISA 16-bit A layout. ----
    v16h aq;
    {
        const _Float16* qrow = QH + (((size_t)b << 10) + (i0 + ln)) * CQK_;
        v8h lo = *(const v8h*)(qrow + kb);
        v8h hh = *(const v8h*)(qrow + kb + 16);
        aq = __builtin_shufflevector(lo, hh, 0, 1, 2, 3, 4, 5, 6, 7,
                                             8, 9, 10, 11, 12, 13, 14, 15);
    }

    // ---- energy: 64 j-tiles over 8 waves, software pipelined ----
    {
        auto loadK = [&](int jt) -> v16h {
            int j = (jt << 4) + ln;
            const _Float16* kr = KH + (((size_t)b << 10) + j) * CQK_ + kb;
            v8h l8 = *(const v8h*)(kr);
            v8h h8 = *(const v8h*)(kr + 16);
            return __builtin_shufflevector(l8, h8, 0, 1, 2, 3, 4, 5, 6, 7,
                                                   8, 9, 10, 11, 12, 13, 14, 15);
        };
        v16h bf = loadK(wave);
        for (int jt = wave; jt < 64; jt += 8) {
            int jn = (jt + 8 < 64) ? jt + 8 : jt;
            v16h bfn = loadK(jn);
            v8f c = {};
            c = __builtin_amdgcn_wmma_f32_16x16x32_f16(false, aq, false, bf,
                                                       (short)0, c, false, false);
#pragma unroll
            for (int r = 0; r < 8; ++r)             // D: M=r+mo (i), N=ln (j)
                P[(r + mo) * PSTR + (jt << 4) + ln] = (_Float16)c[r];
            bf = bfn;
        }
    }
    __syncthreads();

    // ---- softmax over j (1024) per row i; 16 threads/row, vector LDS ----
    {
        int i = t >> 4, ch = t & 15;
        _Float16* row = &P[i * PSTR + (ch << 6)];   // 64 contiguous halves
        float m = -3.0e38f;
#pragma unroll
        for (int k = 0; k < 8; ++k) {
            v8h v = *(const v8h*)(row + (k << 3));
#pragma unroll
            for (int e = 0; e < 8; ++e) m = fmaxf(m, (float)v[e]);
        }
        red[i * 16 + ch] = m; __syncthreads();
        if (ch == 0) {
            float mm = red[i * 16];
            for (int k = 1; k < 16; ++k) mm = fmaxf(mm, red[i * 16 + k]);
            rowmax[i] = mm;
        }
        __syncthreads();
        float M = rowmax[i], s = 0.f;
#pragma unroll
        for (int k = 0; k < 8; ++k) {
            v8h v = *(const v8h*)(row + (k << 3));
#pragma unroll
            for (int e = 0; e < 8; ++e) s += __expf((float)v[e] - M);
        }
        red[i * 16 + ch] = s; __syncthreads();
        if (ch == 0) {
            float ss = red[i * 16];
            for (int k = 1; k < 16; ++k) ss += red[i * 16 + k];
            rowsum[i] = ss;
        }
        __syncthreads();
        float inv = 1.f / rowsum[i];
#pragma unroll
        for (int k = 0; k < 8; ++k) {
            v8h v = *(const v8h*)(row + (k << 3));
            v8h o;
#pragma unroll
            for (int e = 0; e < 8; ++e)
                o[e] = (_Float16)(__expf((float)v[e] - M) * inv);
            *(v8h*)(row + (k << 3)) = o;
        }
    }
    __syncthreads();

    // ---- PV: out[c,i] = sum_j v[c,j]*p[i,j]; pipelined, 2 c-tiles/wave ----
    const float gam = gamma[0];
    {
        const _Float16* prow = &P[ln * PSTR];       // B-fragment row (i = ln)
        int c0 = (wave << 4) + ln;
        int c1 = ((wave + 8) << 4) + ln;
        const _Float16* v0 = VH + (((size_t)b << 8) + c0) * LL;
        const _Float16* v1 = VH + (((size_t)b << 8) + c1) * LL;

        auto loadB = [&](int jc) -> v16h {          // K=j rows, N=i cols
            const _Float16* p = prow + (jc << 5) + kb;
            v8h a = *(const v8h*)(p);
            v8h bb = *(const v8h*)(p + 16);
            return __builtin_shufflevector(a, bb, 0, 1, 2, 3, 4, 5, 6, 7,
                                                  8, 9, 10, 11, 12, 13, 14, 15);
        };
        auto loadA = [&](const _Float16* vp, int jc) -> v16h {
            v8h l = *(const v8h*)(vp + (jc << 5) + kb);
            v8h h = *(const v8h*)(vp + (jc << 5) + kb + 16);
            return __builtin_shufflevector(l, h, 0, 1, 2, 3, 4, 5, 6, 7,
                                                 8, 9, 10, 11, 12, 13, 14, 15);
        };

        v8f acc0 = {}, acc1 = {};
        v16h pb = loadB(0), a0 = loadA(v0, 0), a1 = loadA(v1, 0);
#pragma unroll 2
        for (int jc = 0; jc < 32; ++jc) {
            int jn = (jc + 1 < 32) ? jc + 1 : jc;
            v16h pbn = loadB(jn);
            v16h a0n = loadA(v0, jn);
            v16h a1n = loadA(v1, jn);
            if (jc + 2 < 32) {                      // global_prefetch_b8 on V stream
                __builtin_prefetch(v0 + ((jc + 2) << 5), 0, 1);
                __builtin_prefetch(v1 + ((jc + 2) << 5), 0, 1);
            }
            acc0 = __builtin_amdgcn_wmma_f32_16x16x32_f16(false, a0, false, pb,
                                                          (short)0, acc0, false, false);
            acc1 = __builtin_amdgcn_wmma_f32_16x16x32_f16(false, a1, false, pb,
                                                          (short)0, acc1, false, false);
            pb = pbn; a0 = a0n; a1 = a1n;
        }
#pragma unroll
        for (int r = 0; r < 8; ++r) {
            int cA = (wave << 4) + r + mo;
            int cB = ((wave + 8) << 4) + r + mo;
            size_t iA = (((size_t)b << 8) + cA) * LL + i0 + ln;
            size_t iB = (((size_t)b << 8) + cB) * LL + i0 + ln;
            x1[iA] = gam * acc0[r] + x0[iA];
            x1[iB] = gam * acc1[r] + x0[iB];
        }
    }
}

// ======================================================================
// 7) conv1 (K=256 -> 64 ch) + BN1 stats.  block = (b, 16-col tile)
// ======================================================================
__global__ void __launch_bounds__(256) k_conv1(
        const float* __restrict__ x1, const float* __restrict__ w1,
        const float* __restrict__ b1, float* __restrict__ y1,
        float* __restrict__ sum1, float* __restrict__ sq1) {
    int jt = blockIdx.x & 63, b = blockIdx.x >> 6;
    int j0 = jt << 4;
    int t = threadIdx.x;
    __shared__ float xt[C0C][16];
    __shared__ float ssum[FF], ssq[FF];
    if (t < FF) { ssum[t] = 0.f; ssq[t] = 0.f; }
    {
        const float* src = x1 + (((size_t)b * C0C + t) << 10) + j0;
#pragma unroll
        for (int jj = 0; jj < 16; ++jj) xt[t][jj] = src[jj];
    }
    __syncthreads();
    int f = t & 63, grp = t >> 6;                   // 4 cols per thread
    const float* wr = w1 + (size_t)f * C0C;
    float acc[4];
    float bias = b1[f];
#pragma unroll
    for (int k = 0; k < 4; ++k) acc[k] = bias;
    for (int cc = 0; cc < C0C; ++cc) {
        float w = wr[cc];
#pragma unroll
        for (int k = 0; k < 4; ++k) acc[k] += w * xt[cc][grp * 4 + k];
    }
    float ps = 0.f, pq = 0.f;
    size_t ob = (((size_t)b * FF + f) << 10) + j0 + grp * 4;
#pragma unroll
    for (int k = 0; k < 4; ++k) {
        y1[ob + k] = acc[k];
        ps += acc[k]; pq += acc[k] * acc[k];
    }
    atomicAdd(&ssum[f], ps); atomicAdd(&ssq[f], pq);
    __syncthreads();
    if (t < FF) { atomicAdd(&sum1[t], ssum[t]); atomicAdd(&sq1[t], ssq[t]); }
}

// ======================================================================
// 8) BN1 + ReLU + conv_out fused.  block = (b, 128-col tile)
// ======================================================================
__global__ void __launch_bounds__(256) k_out(
        const float* __restrict__ y1, const float* __restrict__ sc1,
        const float* __restrict__ sh1, const float* __restrict__ wout,
        const float* __restrict__ bout, float* __restrict__ out) {
    int ht = blockIdx.x & 7, b = blockIdx.x >> 3;
    int h0 = ht << 7;
    int t = threadIdx.x;
    __shared__ float a[FF][128];
    for (int rep = 0; rep < 32; ++rep) {
        int idx = t + (rep << 8);
        int f = idx >> 7, x = idx & 127;
        float v = y1[(((size_t)b * FF + f) << 10) + h0 + x];
        a[f][x] = fmaxf(v * sc1[f] + sh1[f], 0.f);
    }
    __syncthreads();
#pragma unroll
    for (int rep = 0; rep < 4; ++rep) {
        int idx = t + (rep << 8);
        int o = idx >> 7, x = idx & 127;
        float acc = bout[o];
        const float* wr = wout + o * FF;
#pragma unroll 8
        for (int f = 0; f < FF; ++f) acc += wr[f] * a[f][x];
        out[(((size_t)b * CIN + o) << 10) + h0 + x] = acc;
    }
}

// ======================================================================
extern "C" void kernel_launch(void* const* d_in, const int* in_sizes, int n_in,
                              void* d_out, int out_size, void* d_ws, size_t ws_size,
                              hipStream_t stream) {
    const float* z     = (const float*)d_in[0];
    const float* w_pre = (const float*)d_in[1];
    const float* b_pre = (const float*)d_in[2];
    const float* w0    = (const float*)d_in[3];
    const float* b0    = (const float*)d_in[4];
    const float* g0    = (const float*)d_in[5];
    const float* be0   = (const float*)d_in[6];
    const float* wq    = (const float*)d_in[7];
    const float* bq    = (const float*)d_in[8];
    const float* wk    = (const float*)d_in[9];
    const float* bk    = (const float*)d_in[10];
    const float* wv    = (const float*)d_in[11];
    const float* bv    = (const float*)d_in[12];
    const float* gamma = (const float*)d_in[13];
    const float* w1    = (const float*)d_in[14];
    const float* b1    = (const float*)d_in[15];
    const float* g1    = (const float*)d_in[16];
    const float* be1   = (const float*)d_in[17];
    const float* w_out = (const float*)d_in[18];
    const float* b_out = (const float*)d_in[19];
    float* out = (float*)d_out;

    char* ws = (char*)d_ws;
    float*     XPRE = (float*)(ws + OFF_XPRE);
    float*     X0   = (float*)(ws + OFF_X0);
    float*     X1   = (float*)(ws + OFF_X1);
    float*     Y1   = (float*)(ws + OFF_Y1);
    _Float16*  QH   = (_Float16*)(ws + OFF_QH);
    _Float16*  KH   = (_Float16*)(ws + OFF_KH);
    _Float16*  VH   = (_Float16*)(ws + OFF_VH);
    float* STATS = (float*)(ws + OFF_STATS);
    float* sum0 = STATS;            // 256
    float* sq0  = STATS + 256;      // 256
    float* sum1 = STATS + 512;      // 64
    float* sq1  = STATS + 576;      // 64
    float* sc0  = STATS + 640;      // 256
    float* sh0  = STATS + 896;      // 256
    float* sc1  = STATS + 1152;     // 64
    float* sh1  = STATS + 1216;     // 64

    hipMemsetAsync(STATS, 0, 640 * sizeof(float), stream);   // zero sums/sumsqs

    k_pre   <<<2048, 256, 0, stream>>>(z, w_pre, b_pre, XPRE);
    k_conv0 <<<BB * C0C, 256, 0, stream>>>(XPRE, w0, b0, X0, sum0, sq0);
    k_bnfin <<<1, 256, 0, stream>>>(sum0, sq0, g0, be0, sc0, sh0, C0C,
                                    1.f / (float)(BB * LL));
    k_bnrelu<<<16384, 256, 0, stream>>>(X0, sc0, sh0);
    k_qkv   <<<BB * (LL / 16), 256, 0, stream>>>(X0, wq, bq, wk, bk, wv, bv,
                                                 QH, KH, VH);
    k_attn  <<<BB * (LL / 16), 256, 0, stream>>>(QH, KH, VH, X0, gamma, X1);
    k_conv1 <<<BB * (LL / 16), 256, 0, stream>>>(X1, w1, b1, Y1, sum1, sq1);
    k_bnfin <<<1, 64, 0, stream>>>(sum1, sq1, g1, be1, sc1, sh1, FF,
                                   1.f / (float)(BB * LL));
    k_out   <<<BB * (LL / 128), 256, 0, stream>>>(Y1, sc1, sh1, w_out, b_out, out);
}
